// ToBEVConvolution_13194139533436
// MI455X (gfx1250) — matrix operand order repaired and checked by reading, not compiled, not run on hardware
//
#include <hip/hip_runtime.h>

#define IN_C   16
#define OUT_C  16
#define NKER   8
#define GY     512
#define GB     4

typedef __attribute__((ext_vector_type(16))) __bf16          v16bf;
typedef __attribute__((ext_vector_type(16))) unsigned short  v16u;
typedef __attribute__((ext_vector_type(8)))  float           v8f;

// float -> bf16 bits, round-to-nearest-even
__device__ __forceinline__ unsigned short f2bf(float x) {
    unsigned u = __builtin_bit_cast(unsigned, x);
    unsigned r = u + 0x7FFFu + ((u >> 16) & 1u);
    return (unsigned short)(r >> 16);
}

// One wave processes 16 points per tile:
//   D(16x16) = A'(16x128 bf16, kidx-masked feats) x W'(128x16 bf16, all 8 kernels)
// as 4x v_wmma_f32_16x16x32_bf16, then 16x16 fp32 atomic scatter into the BEV grid.
__global__ __launch_bounds__(256) void tobev_wmma_kernel(
    const float* __restrict__ feats,     // [N,16] f32
    const float* __restrict__ W,         // [8,16,16] f32  == W'[128][16]
    const int*   __restrict__ coords,    // [N,4] i32 (x, z(height), y, b)
    const int*   __restrict__ stridep,   // [1]
    float*       __restrict__ out,       // [512*512*4, 16] f32 (pre-zeroed)
    int n)
{
    const int lane   = threadIdx.x & 31;
    const int h      = lane >> 4;        // K-half selector for A/B layouts
    const int col    = lane & 15;        // matrix column N / row M=lane%16
    const int wave   = (int)((blockIdx.x * blockDim.x + threadIdx.x) >> 5);
    const int nwaves = (int)((gridDim.x * blockDim.x) >> 5);
    const int stride = stridep[0];

    // ---- Hoist B: all 8 kernels as a 128x16 bf16 matrix, 4 chunks of K=32 ----
    // B layout: lane L, element j  ->  K = 32*c + 16*(j/8) + 8*(L/16) + j%8, N = L%16
    v16bf bm[4];
#pragma unroll
    for (int c = 0; c < 4; ++c) {
        v16u bu = {};
#pragma unroll
        for (int j = 0; j < 16; ++j) {
            int kk = 32 * c + 16 * (j >> 3) + 8 * h + (j & 7);
            bu[j] = f2bf(W[kk * OUT_C + col]);
        }
        bm[c] = __builtin_bit_cast(v16bf, bu);
    }

    const int4*   coords4 = (const int4*)coords;
    const float4* feats4  = (const float4*)feats;
    const int     ntiles  = (n + 15) >> 4;

    for (int t = wave; t < ntiles; t += nwaves) {
        const int p = t * 16 + col;      // this lane's point (row M = lane%16)

        int    km = 0, lin = 0;
        float4 f0 = {0.f, 0.f, 0.f, 0.f};
        float4 f1 = {0.f, 0.f, 0.f, 0.f};
        if (p < n) {
            int4 crd = coords4[p];                       // global_load_b128
            km  = crd.y / stride;                        // kernel select
            lin = (crd.x * GY + crd.z) * GB + crd.w;     // BEV cell
            // A-layout only ever needs feats[8h .. 8h+7] in this lane:
            f0 = feats4[p * 4 + 2 * h];                  // global_load_b128
            f1 = feats4[p * 4 + 2 * h + 1];              // global_load_b128
        }
        unsigned short fb[8];
        fb[0] = f2bf(f0.x); fb[1] = f2bf(f0.y); fb[2] = f2bf(f0.z); fb[3] = f2bf(f0.w);
        fb[4] = f2bf(f1.x); fb[5] = f2bf(f1.y); fb[6] = f2bf(f1.z); fb[7] = f2bf(f1.w);

        // ---- 4 chained WMMAs over K = 0..127 ----
        v8f acc = {};
#pragma unroll
        for (int c = 0; c < 4; ++c) {
            // A chunk c: row M's nonzeros live at K in [16*km, 16*km+16).
            // chunk index = km>>1, upper/lower 16 of the chunk = km&1.
            v16u au = {};
#pragma unroll
            for (int j = 0; j < 16; ++j) {
                bool on = ((km >> 1) == c) & ((km & 1) == (j >> 3));
                au[j] = on ? fb[j & 7] : (unsigned short)0;
            }
            acc = __builtin_amdgcn_wmma_f32_16x16x32_bf16(
                      /*neg_a=*/false, __builtin_bit_cast(v16bf, au),
                      /*neg_b=*/false, bm[c],
                      /*c_mod=*/(short)0, acc,
                      /*reuse_a=*/false, /*reuse_b=*/false);
        }

        // ---- Scatter: C element r -> point M = r + 8h, channel N = col ----
#pragma unroll
        for (int r = 0; r < 8; ++r) {
            int mm   = r + 8 * h;
            int linp = __shfl(lin, mm, 32);              // ds_bpermute
            unsafeAtomicAdd(out + linp * OUT_C + col, acc[r]);
        }
    }
}

extern "C" void kernel_launch(void* const* d_in, const int* in_sizes, int n_in,
                              void* d_out, int out_size, void* d_ws, size_t ws_size,
                              hipStream_t stream) {
    (void)n_in; (void)d_ws; (void)ws_size;
    const float* feats   = (const float*)d_in[0];
    const float* W       = (const float*)d_in[1];
    const int*   coords  = (const int*)d_in[2];
    const int*   stridep = (const int*)d_in[3];
    float*       out     = (float*)d_out;
    const int    n       = in_sizes[0] / IN_C;

    // Harness poisons d_out; grid must start at zero for the scatter-sum.
    hipMemsetAsync(d_out, 0, (size_t)out_size * sizeof(float), stream);

    tobev_wmma_kernel<<<dim3(512), dim3(256), 0, stream>>>(
        feats, W, coords, stridep, out, n);
}
